// Net_36610301231246
// MI455X (gfx1250) — matrix operand order repaired and checked by reading, not compiled, not run on hardware
//
#include <hip/hip_runtime.h>
#include <hip/hip_bf16.h>
#include <math.h>

typedef __attribute__((ext_vector_type(16))) _Float16 v16h;
typedef __attribute__((ext_vector_type(8)))  float    v8f;

#define D_FEAT   128
#define N_GAUSS  32
#define LN2F     0.69314718055994531f
#define CUTOFF_F 5.0f

__device__ __forceinline__ float ssp_f(float x) {
  // softplus(x) - ln2, numerically stable
  return fmaxf(x, 0.0f) + log1pf(expf(-fabsf(x))) - LN2F;
}

__device__ __forceinline__ v8f wmma_f16(v16h a, v16h b, v8f c) {
  return __builtin_amdgcn_wmma_f32_16x16x32_f16(false, a, false, b, (short)0, c,
                                                false, false);
}

// B-tile (32x16) loader from transposed f16 weights wT[N][K]:
// lane L -> column n0+(L&15); K rows k0 + (L<16 ? 0..7,16..23 : 8..15,24..31).
__device__ __forceinline__ v16h load_bT(const _Float16* __restrict__ wT, int ldk,
                                        int n0, int k0, int lane) {
  const int n  = n0 + (lane & 15);
  const int kb = k0 + ((lane < 16) ? 0 : 8);
  const _Float16* p = wT + (size_t)n * ldk + kb;
  v16h b;
#pragma unroll
  for (int s = 0; s < 16; ++s) b[s] = p[s + ((s >= 8) ? 8 : 0)];
  return b;
}

// Load all 4 K-step B tiles (K=128) for one 16-column chunk.
__device__ __forceinline__ void load_b4(v16h* __restrict__ dst,
                                        const _Float16* __restrict__ wT,
                                        int n0, int lane) {
#pragma unroll
  for (int kk = 0; kk < 4; ++kk) dst[kk] = load_bT(wT, D_FEAT, n0, kk * 32, lane);
}

// ---------------------------------------------------------------- utility
__global__ void zero_f32_kernel(float* __restrict__ p, int n) {
  int i = blockIdx.x * blockDim.x + threadIdx.x;
  if (i < n) p[i] = 0.0f;
}

// src: [T][K][N] f32 (row-major)  ->  dst: [T][N][K] f16
__global__ void transpose_to_f16_kernel(const float* __restrict__ src,
                                        _Float16* __restrict__ dst,
                                        int K, int N, int total) {
  int idx = blockIdx.x * blockDim.x + threadIdx.x;
  if (idx >= total) return;
  int kn  = K * N;
  int t   = idx / kn;
  int rem = idx - t * kn;
  int k   = rem / N;
  int n   = rem - k * N;
  dst[(size_t)t * kn + (size_t)n * K + k] = (_Float16)src[idx];
}

__global__ void gather_embed_kernel(const int* __restrict__ r,
                                    const float* __restrict__ embed,
                                    float* __restrict__ h,
                                    _Float16* __restrict__ h16, int total) {
  int idx = blockIdx.x * blockDim.x + threadIdx.x;
  if (idx >= total) return;
  int atom = idx >> 7;
  int ch   = idx & (D_FEAT - 1);
  float v  = embed[(size_t)r[atom] * D_FEAT + ch];
  h[idx]   = v;
  h16[idx] = (_Float16)v;
}

// -------------------------------------------------- node input GEMM (WMMA)
// hf = h16 @ inw + inb   (out f32 [M][128]); one wave per 16-row tile.
__global__ __launch_bounds__(64)
void node_in_kernel(const _Float16* __restrict__ h16,
                    const _Float16* __restrict__ wT,
                    const float* __restrict__ bias,
                    float* __restrict__ out, int M) {
  const int wave = threadIdx.x >> 5;
  const int lane = threadIdx.x & 31;
  const int tile = blockIdx.x * 2 + wave;
  const int m0   = tile * 16;
  if (m0 >= M) return;
  const int m   = lane & 15;
  const int kbA = (lane < 16) ? 0 : 8;
  const _Float16* arow = h16 + (size_t)(m0 + m) * D_FEAT;
  v16h Ak[4];
#pragma unroll
  for (int kk = 0; kk < 4; ++kk) {
    const _Float16* pa = arow + kk * 32 + kbA;
#pragma unroll
    for (int s = 0; s < 8; ++s) { Ak[kk][s] = pa[s]; Ak[kk][s + 8] = pa[s + 16]; }
  }
  v16h B[2][4];
  load_b4(B[0], wT, 0, lane);
#pragma unroll
  for (int c = 0; c < 8; ++c) {
    if (c < 7) load_b4(B[(c + 1) & 1], wT, (c + 1) * 16, lane);
    v8f acc = {};  // inline-0 C operand
#pragma unroll
    for (int kk = 0; kk < 4; ++kk) acc = wmma_f16(Ak[kk], B[c & 1][kk], acc);
    const float bv = bias[c * 16 + m];
#pragma unroll
    for (int v = 0; v < 8; ++v) {
      int row = (lane < 16) ? v : (v + 8);
      out[(size_t)(m0 + row) * D_FEAT + c * 16 + m] = acc[v] + bv;
    }
  }
}

// ------------------------- fused edge filter MLP (WMMA) + scatter-aggregate
__global__ __launch_bounds__(128)
void edge_filter_agg_kernel(const float* __restrict__ xyz,
                            const int* __restrict__ aidx,
                            const _Float16* __restrict__ fw1T,
                            const float* __restrict__ fb1,
                            const _Float16* __restrict__ fw2T,
                            const float* __restrict__ fb2,
                            const float* __restrict__ hf,
                            float* __restrict__ agg, int n_edges) {
  __shared__ _Float16 lds_hid[4][16 * D_FEAT];
  __shared__ float    lds_w  [4][16 * D_FEAT];
  const int wave = threadIdx.x >> 5;
  const int lane = threadIdx.x & 31;
  const int tile = blockIdx.x * 4 + wave;
  const int e0   = tile * 16;
  const bool valid = (e0 < n_edges);
  const int m = lane & 15;

  int E = e0 + m;
  if (!valid || E >= n_edges) E = 0;  // clamp; garbage math is discarded
  const int ia = aidx[2 * E], ja = aidx[2 * E + 1];
  const float dx = xyz[3 * ia + 0] - xyz[3 * ja + 0];
  const float dy = xyz[3 * ia + 1] - xyz[3 * ja + 1];
  const float dz = xyz[3 * ia + 2] - xyz[3 * ja + 2];
  const float d  = sqrtf(dx * dx + dy * dy + dz * dz);

  const float width = CUTOFF_F / (float)(N_GAUSS - 1);
  const float coeff = -0.5f / (width * width);
  const int kbA = (lane < 16) ? 0 : 8;
  v16h A;
#pragma unroll
  for (int s = 0; s < 16; ++s) {
    int g   = kbA + s + ((s >= 8) ? 8 : 0);
    float t = d - width * (float)g;
    A[s] = (_Float16)expf(coeff * t * t);
  }

  // layer 1: K = 32 -> single WMMA per column chunk, B double-buffered
  {
    v16h B1[2];
    B1[0] = load_bT(fw1T, N_GAUSS, 0, 0, lane);
#pragma unroll
    for (int c = 0; c < 8; ++c) {
      if (c < 7) B1[(c + 1) & 1] = load_bT(fw1T, N_GAUSS, (c + 1) * 16, 0, lane);
      v8f acc = {};
      acc = wmma_f16(A, B1[c & 1], acc);
      const float bv = fb1[c * 16 + m];
#pragma unroll
      for (int v = 0; v < 8; ++v) {
        int row = (lane < 16) ? v : (v + 8);
        lds_hid[wave][row * D_FEAT + c * 16 + m] = (_Float16)ssp_f(acc[v] + bv);
      }
    }
  }
  __syncthreads();

  // layer 2: K = 128; hoist A tiles (c-invariant), double-buffer B
  {
    v16h Ah[4];
#pragma unroll
    for (int kk = 0; kk < 4; ++kk) {
      const _Float16* pa = &lds_hid[wave][m * D_FEAT + kk * 32 + kbA];
#pragma unroll
      for (int s = 0; s < 8; ++s) { Ah[kk][s] = pa[s]; Ah[kk][s + 8] = pa[s + 16]; }
    }
    v16h B[2][4];
    load_b4(B[0], fw2T, 0, lane);
#pragma unroll
    for (int c = 0; c < 8; ++c) {
      if (c < 7) load_b4(B[(c + 1) & 1], fw2T, (c + 1) * 16, lane);
      v8f acc = {};
#pragma unroll
      for (int kk = 0; kk < 4; ++kk) acc = wmma_f16(Ah[kk], B[c & 1][kk], acc);
      const float bv = fb2[c * 16 + m];
#pragma unroll
      for (int v = 0; v < 8; ++v) {
        int row = (lane < 16) ? v : (v + 8);
        lds_w[wave][row * D_FEAT + c * 16 + m] = acc[v] + bv;
      }
    }
  }
  __syncthreads();
  if (!valid) return;

  // scatter: 128 channels = 32 lanes x 4
#pragma unroll 1
  for (int e = 0; e < 16; ++e) {
    const int E2 = e0 + e;
    if (E2 >= n_edges) break;
    const int ii = aidx[2 * E2], jj = aidx[2 * E2 + 1];
    const float4 w4 = *(const float4*)&lds_w[wave][e * D_FEAT + lane * 4];
    const float4 hj = *(const float4*)&hf[(size_t)jj * D_FEAT + lane * 4];
    const float4 hi = *(const float4*)&hf[(size_t)ii * D_FEAT + lane * 4];
    float* ai = &agg[(size_t)ii * D_FEAT + lane * 4];
    float* aj = &agg[(size_t)jj * D_FEAT + lane * 4];
    atomicAdd(ai + 0, hj.x * w4.x); atomicAdd(ai + 1, hj.y * w4.y);
    atomicAdd(ai + 2, hj.z * w4.z); atomicAdd(ai + 3, hj.w * w4.w);
    atomicAdd(aj + 0, hi.x * w4.x); atomicAdd(aj + 1, hi.y * w4.y);
    atomicAdd(aj + 2, hi.z * w4.z); atomicAdd(aj + 3, hi.w * w4.w);
  }
}

// ------------------- node output MLP (WMMA): h += ssp(agg@ow1+ob1)@ow2+ob2
__global__ __launch_bounds__(64)
void node_out_kernel(const float* __restrict__ agg,
                     const _Float16* __restrict__ w1T, const float* __restrict__ b1,
                     const _Float16* __restrict__ w2T, const float* __restrict__ b2,
                     float* __restrict__ h, _Float16* __restrict__ h16, int M) {
  __shared__ _Float16 lds_u[2][16 * D_FEAT];
  const int wave = threadIdx.x >> 5;
  const int lane = threadIdx.x & 31;
  const int tile = blockIdx.x * 2 + wave;
  const int m0   = tile * 16;
  const bool valid = (m0 < M);
  const int m   = lane & 15;
  const int kbA = (lane < 16) ? 0 : 8;
  const size_t mrow = valid ? (size_t)(m0 + m) * D_FEAT : 0;

  // layer 1
  {
    v16h Ak[4];
#pragma unroll
    for (int kk = 0; kk < 4; ++kk) {
      const float* pa = agg + mrow + kk * 32 + kbA;
#pragma unroll
      for (int s = 0; s < 8; ++s) {
        Ak[kk][s]     = (_Float16)pa[s];
        Ak[kk][s + 8] = (_Float16)pa[s + 16];
      }
    }
    v16h B[2][4];
    load_b4(B[0], w1T, 0, lane);
#pragma unroll
    for (int c = 0; c < 8; ++c) {
      if (c < 7) load_b4(B[(c + 1) & 1], w1T, (c + 1) * 16, lane);
      v8f acc = {};
#pragma unroll
      for (int kk = 0; kk < 4; ++kk) acc = wmma_f16(Ak[kk], B[c & 1][kk], acc);
      const float bv = b1[c * 16 + m];
#pragma unroll
      for (int v = 0; v < 8; ++v) {
        int row = (lane < 16) ? v : (v + 8);
        lds_u[wave][row * D_FEAT + c * 16 + m] = (_Float16)ssp_f(acc[v] + bv);
      }
    }
  }
  __syncthreads();
  // layer 2
  {
    v16h Ah[4];
#pragma unroll
    for (int kk = 0; kk < 4; ++kk) {
      const _Float16* pa = &lds_u[wave][m * D_FEAT + kk * 32 + kbA];
#pragma unroll
      for (int s = 0; s < 8; ++s) { Ah[kk][s] = pa[s]; Ah[kk][s + 8] = pa[s + 16]; }
    }
    v16h B[2][4];
    load_b4(B[0], w2T, 0, lane);
#pragma unroll
    for (int c = 0; c < 8; ++c) {
      if (c < 7) load_b4(B[(c + 1) & 1], w2T, (c + 1) * 16, lane);
      v8f acc = {};
#pragma unroll
      for (int kk = 0; kk < 4; ++kk) acc = wmma_f16(Ah[kk], B[c & 1][kk], acc);
      if (valid) {
        const float bv = b2[c * 16 + m];
#pragma unroll
        for (int v = 0; v < 8; ++v) {
          int row = (lane < 16) ? v : (v + 8);
          size_t idx = (size_t)(m0 + row) * D_FEAT + c * 16 + m;
          float hv = h[idx] + acc[v] + bv;
          h[idx]   = hv;
          h16[idx] = (_Float16)hv;
        }
      }
    }
  }
}

// ------------------------------------------- readout MLP + per-mol reduce
__global__ __launch_bounds__(128)
void readout_kernel(const float* __restrict__ h, const float* __restrict__ aw1,
                    const float* __restrict__ ab1, const float* __restrict__ aw2,
                    const float* __restrict__ ab2, const int* __restrict__ mol_idx,
                    float* __restrict__ out, int M) {
  const int wave = threadIdx.x >> 5;
  const int lane = threadIdx.x & 31;
  const int atom = blockIdx.x * 4 + wave;
  if (atom >= M) return;
  const float* hr = h + (size_t)atom * D_FEAT;
  float s0 = ab1[lane], s1 = ab1[lane + 32];
#pragma unroll 4
  for (int k = 0; k < D_FEAT; ++k) {
    float hk = hr[k];
    s0 = fmaf(hk, aw1[k * 64 + lane], s0);
    s1 = fmaf(hk, aw1[k * 64 + lane + 32], s1);
  }
  float p = ssp_f(s0) * aw2[lane] + ssp_f(s1) * aw2[lane + 32];
#pragma unroll
  for (int offm = 16; offm > 0; offm >>= 1) p += __shfl_xor(p, offm, 32);
  if (lane == 0) atomicAdd(&out[mol_idx[atom]], p + ab2[0]);
}

__global__ void bond_kernel(const float* __restrict__ xyz,
                            const int* __restrict__ badj,
                            const float* __restrict__ blen,
                            const int* __restrict__ mol_idx,
                            float* __restrict__ out, int NB) {
  int b = blockIdx.x * blockDim.x + threadIdx.x;
  if (b >= NB) return;
  int b0 = badj[2 * b], b1 = badj[2 * b + 1];
  float dx = xyz[3 * b0 + 0] - xyz[3 * b1 + 0];
  float dy = xyz[3 * b0 + 1] - xyz[3 * b1 + 1];
  float dz = xyz[3 * b0 + 2] - xyz[3 * b1 + 2];
  float bd = sqrtf(dx * dx + dy * dy + dz * dz);
  float t  = bd - blen[b];
  atomicAdd(&out[mol_idx[b0]], 50.0f * t * t);
}

// -------------------------------------------------------------------- host
extern "C" void kernel_launch(void* const* d_in, const int* in_sizes, int n_in,
                              void* d_out, int out_size, void* d_ws, size_t ws_size,
                              hipStream_t stream) {
  const int*   r        = (const int*)  d_in[0];
  const float* xyz      = (const float*)d_in[1];
  const int*   aidx     = (const int*)  d_in[2];
  const int*   mol_idx  = (const int*)  d_in[3];
  const int*   bond_adj = (const int*)  d_in[4];
  const float* bond_len = (const float*)d_in[5];
  const float* embed    = (const float*)d_in[6];
  const float* fw1      = (const float*)d_in[7];
  const float* fb1      = (const float*)d_in[8];
  const float* fw2      = (const float*)d_in[9];
  const float* fb2      = (const float*)d_in[10];
  const float* inw      = (const float*)d_in[11];
  const float* inb      = (const float*)d_in[12];
  const float* ow1      = (const float*)d_in[13];
  const float* ob1      = (const float*)d_in[14];
  const float* ow2      = (const float*)d_in[15];
  const float* ob2      = (const float*)d_in[16];
  const float* aw1      = (const float*)d_in[17];
  const float* ab1      = (const float*)d_in[18];
  const float* aw2      = (const float*)d_in[19];
  const float* ab2      = (const float*)d_in[20];

  const int n_atoms = in_sizes[0];
  const int n_edges = in_sizes[2] / 2;
  const int n_bonds = in_sizes[4] / 2;
  const int T       = in_sizes[8] / D_FEAT;

  // workspace carving (256B aligned regions)
  char* ws = (char*)d_ws;
  size_t off = 0;
  auto carve = [&](size_t bytes) -> char* {
    char* p = ws + off;
    off = (off + bytes + 255) & ~(size_t)255;
    return p;
  };
  _Float16* fw1T = (_Float16*)carve((size_t)T * D_FEAT * N_GAUSS * sizeof(_Float16));
  _Float16* fw2T = (_Float16*)carve((size_t)T * D_FEAT * D_FEAT * sizeof(_Float16));
  _Float16* inwT = (_Float16*)carve((size_t)T * D_FEAT * D_FEAT * sizeof(_Float16));
  _Float16* ow1T = (_Float16*)carve((size_t)T * D_FEAT * D_FEAT * sizeof(_Float16));
  _Float16* ow2T = (_Float16*)carve((size_t)T * D_FEAT * D_FEAT * sizeof(_Float16));
  float*    h    = (float*)   carve((size_t)n_atoms * D_FEAT * sizeof(float));
  _Float16* h16  = (_Float16*)carve((size_t)n_atoms * D_FEAT * sizeof(_Float16));
  float*    hf   = (float*)   carve((size_t)n_atoms * D_FEAT * sizeof(float));
  float*    agg  = (float*)   carve((size_t)n_atoms * D_FEAT * sizeof(float));
  (void)ws_size;

  float* out = (float*)d_out;
  zero_f32_kernel<<<(out_size + 255) / 256, 256, 0, stream>>>(out, out_size);

  // transpose/convert weights to f16 [N][K]
  {
    int tot1 = T * N_GAUSS * D_FEAT;
    transpose_to_f16_kernel<<<(tot1 + 255) / 256, 256, 0, stream>>>(fw1, fw1T, N_GAUSS, D_FEAT, tot1);
    int tot2 = T * D_FEAT * D_FEAT;
    transpose_to_f16_kernel<<<(tot2 + 255) / 256, 256, 0, stream>>>(fw2, fw2T, D_FEAT, D_FEAT, tot2);
    transpose_to_f16_kernel<<<(tot2 + 255) / 256, 256, 0, stream>>>(inw, inwT, D_FEAT, D_FEAT, tot2);
    transpose_to_f16_kernel<<<(tot2 + 255) / 256, 256, 0, stream>>>(ow1, ow1T, D_FEAT, D_FEAT, tot2);
    transpose_to_f16_kernel<<<(tot2 + 255) / 256, 256, 0, stream>>>(ow2, ow2T, D_FEAT, D_FEAT, tot2);
  }

  const int htot = n_atoms * D_FEAT;
  gather_embed_kernel<<<(htot + 255) / 256, 256, 0, stream>>>(r, embed, h, h16, htot);

  const int n_tiles = (n_atoms + 15) / 16;
  const int e_tiles = (n_edges + 15) / 16;
  for (int t = 0; t < T; ++t) {
    node_in_kernel<<<(n_tiles + 1) / 2, 64, 0, stream>>>(
        h16, inwT + (size_t)t * D_FEAT * D_FEAT, inb + t * D_FEAT, hf, n_atoms);
    zero_f32_kernel<<<(htot + 255) / 256, 256, 0, stream>>>(agg, htot);
    edge_filter_agg_kernel<<<(e_tiles + 3) / 4, 128, 0, stream>>>(
        xyz, aidx,
        fw1T + (size_t)t * D_FEAT * N_GAUSS, fb1 + t * D_FEAT,
        fw2T + (size_t)t * D_FEAT * D_FEAT,  fb2 + t * D_FEAT,
        hf, agg, n_edges);
    node_out_kernel<<<(n_tiles + 1) / 2, 64, 0, stream>>>(
        agg,
        ow1T + (size_t)t * D_FEAT * D_FEAT, ob1 + t * D_FEAT,
        ow2T + (size_t)t * D_FEAT * D_FEAT, ob2 + t * D_FEAT,
        h, h16, n_atoms);
  }

  readout_kernel<<<(n_atoms + 3) / 4, 128, 0, stream>>>(h, aw1, ab1, aw2, ab2,
                                                        mol_idx, out, n_atoms);
  bond_kernel<<<(n_bonds + 255) / 256, 256, 0, stream>>>(xyz, bond_adj, bond_len,
                                                         mol_idx, out, n_bonds);
}
